// DisAttention_78812649882219
// MI455X (gfx1250) — matrix operand order repaired and checked by reading, not compiled, not run on hardware
//
#include <hip/hip_runtime.h>
#include <hip/hip_bf16.h>

// ---------------------------------------------------------------------------
// DisAttention on MI455X (gfx1250).
// bf16 WMMA everywhere (v_wmma_f32_16x16x32_bf16), f32 accumulate.
// Big GEMMs: block-cooperative, double-buffered LDS staging via
// global_load_async_to_lds_b128 (ASYNCcnt), 8 waves x (64x32) register tiles.
// Top-k: exact 2-level 16-bit radix select on positive-float bit patterns.
// ---------------------------------------------------------------------------

#define BN 8
#define SN 1024
#define DN 768
#define K_TOPK 104857u /* int(1024*1024*0.1) */

typedef __attribute__((ext_vector_type(16))) __bf16 v16bf;
typedef __attribute__((ext_vector_type(8)))  float  v8f;

// ---- bf16 helpers ---------------------------------------------------------
__device__ __forceinline__ unsigned short f32_to_bf16(float f) {
    unsigned u = __float_as_uint(f);
    return (unsigned short)((u + 0x7FFFu + ((u >> 16) & 1u)) >> 16); // RNE
}
__device__ __forceinline__ __bf16 bf_from_bits(unsigned short s) {
    __bf16 r; __builtin_memcpy(&r, &s, 2); return r;
}
__device__ __forceinline__ v16bf frag_from_chunks(uint4 c0, uint4 c1) {
    union { v16bf v; uint4 q[2]; } u;
    u.q[0] = c0; u.q[1] = c1;
    return u.v;
}
__device__ __forceinline__ v16bf frag_from_f32x16(float4 a, float4 b, float4 c, float4 d) {
    float f[16] = {a.x,a.y,a.z,a.w, b.x,b.y,b.z,b.w, c.x,c.y,c.z,c.w, d.x,d.y,d.z,d.w};
    v16bf v;
#pragma unroll
    for (int i = 0; i < 16; ++i) v[i] = bf_from_bits(f32_to_bf16(f[i]));
    return v;
}

// ---- CDNA5 async global->LDS copy (tracked by ASYNCcnt) -------------------
__device__ __forceinline__ void async_copy_b128(unsigned ldsOff, const void* gaddr) {
    asm volatile("global_load_async_to_lds_b128 %0, %1, off"
                 :: "v"(ldsOff), "v"(gaddr) : "memory");
}
__device__ __forceinline__ void wait_async0() {
    asm volatile("s_wait_asynccnt 0x0" ::: "memory");
}
__device__ __forceinline__ unsigned lds_off(const void* p) {
    // LDS flat addresses: low 32 bits are the LDS byte offset (ISA 10.2 aperture)
    return (unsigned)(size_t)p;
}

// ---------------------------------------------------------------------------
// Unified bf16 block GEMM:  C[b] = scale * A[b](MxK) * B[b]^T-ish
//   A element (m,k) at A[b*sA + m*lda + k]      (bf16, K-contiguous)
//   B element (k,n) at B[b*sB + n*ldb + k]      (bf16, K-contiguous)
//   C element (m,n) at C[b*sC + m*ldc + n]      (f32)
// Block: 128x128 macro tile, 8 waves (2x4), each wave 64x32 (4x2 fragments).
// K-step 32: stage A-tile(128x32) + B-tile(128x32) into LDS (rows padded to
// 80B: no 64-bank conflicts, 16B aligned), double buffered, async copies.
// ---------------------------------------------------------------------------
#define ROWB 80      // 64B data + 16B pad
#define TILEB (128 * ROWB)   // 10240 B per matrix tile
#define BUFB (2 * TILEB)     // 20480 B per stage (A+B)

__global__ void __launch_bounds__(256)
bf16_gemm_kernel(const unsigned short* __restrict__ A,
                 const unsigned short* __restrict__ Bm,
                 float* __restrict__ C,
                 int lda, int ldb, int ldc, int K,
                 int Mt, int Nt,
                 long long sA, long long sB, long long sC,
                 float scale) {
    __shared__ __align__(16) unsigned char smem[2 * BUFB]; // 40 KB

    const int t = threadIdx.x;
    const int per = Mt * Nt;
    const int b  = blockIdx.x / per;
    const int tt = blockIdx.x % per;
    const int bm = tt / Nt, bn = tt % Nt;

    const unsigned short* Ab = A + (size_t)b * sA + (size_t)(bm * 128) * lda;
    const unsigned short* Bb = Bm + (size_t)b * sB + (size_t)(bn * 128) * ldb;

    const int wave = t >> 5;
    const int wm = wave >> 2;          // 0..1 : M offset wm*64
    const int wn = wave & 3;           // 0..3 : N offset wn*32
    const int lane = t & 31;
    const int h  = lane >> 4;
    const int mn = lane & 15;

    v8f acc[4][2];
#pragma unroll
    for (int f = 0; f < 4; ++f)
#pragma unroll
        for (int g = 0; g < 2; ++g) acc[f][g] = (v8f){};

    const int nK = K / 32;

    // stage(ks) into buffer ks&1 : 512 16B-chunks per matrix, 2 per thread
    auto stage = [&](int ks) {
        unsigned ldsA = lds_off(smem) + (unsigned)((ks & 1) * BUFB);
        unsigned ldsB = ldsA + TILEB;
        const unsigned short* gA = Ab + ks * 32;
        const unsigned short* gB = Bb + ks * 32;
#pragma unroll
        for (int j = 0; j < 2; ++j) {
            const int ci = j * 256 + t, row = ci >> 1, half = ci & 1;
            async_copy_b128(ldsA + row * ROWB + half * 16,
                            gA + (size_t)row * lda + half * 8);
            async_copy_b128(ldsB + row * ROWB + half * 16,
                            gB + (size_t)row * ldb + half * 8);
        }
    };

    stage(0);
    for (int ks = 0; ks < nK; ++ks) {
        wait_async0();
        __syncthreads();              // stage(ks) visible; compute(ks-1) done
        if (ks + 1 < nK) stage(ks + 1);

        const unsigned char* bufA = smem + (ks & 1) * BUFB;
        const unsigned char* bufB = bufA + TILEB;

        v16bf af[4], bfr[2];
#pragma unroll
        for (int f = 0; f < 4; ++f) {
            const unsigned char* p = bufA + (wm * 64 + f * 16 + mn) * ROWB + 16 * h;
            af[f] = frag_from_chunks(*(const uint4*)p, *(const uint4*)(p + 32));
        }
#pragma unroll
        for (int g = 0; g < 2; ++g) {
            const unsigned char* p = bufB + (wn * 32 + g * 16 + mn) * ROWB + 16 * h;
            bfr[g] = frag_from_chunks(*(const uint4*)p, *(const uint4*)(p + 32));
        }
#pragma unroll
        for (int f = 0; f < 4; ++f)
#pragma unroll
            for (int g = 0; g < 2; ++g)
                acc[f][g] = __builtin_amdgcn_wmma_f32_16x16x32_bf16(
                    false, af[f], false, bfr[g], (short)0, acc[f][g], false, false);
    }

    float* Cb = C + (size_t)b * sC;
#pragma unroll
    for (int f = 0; f < 4; ++f)
#pragma unroll
        for (int g = 0; g < 2; ++g) {
            const int col = bn * 128 + wn * 32 + g * 16 + mn;
#pragma unroll
            for (int r = 0; r < 8; ++r) {
                const int row = bm * 128 + wm * 64 + f * 16 + r + 8 * h;
                Cb[(size_t)row * ldc + col] = acc[f][g][r] * scale;
            }
        }
}

// ---------------------------------------------------------------------------
// QKV projection: Y[8192,768] = X @ W^T + bias -> bf16 (optionally transposed
// per batch for V: Yt[b][n][s]).  Wave computes 16x64 (1 A frag, 4 B frags).
// A (f32): lane runs at k0+8h and k0+16+8h (two float4 pairs).
// B(k,n)=W[n,k]: lane runs at k0+8h and k0+16+8h (two float4 pairs).
// tiles: 512 M x 12 N-blocks = 6144 waves -> 768 blocks
// ---------------------------------------------------------------------------
template <bool TRANSPOSE>
__global__ void __launch_bounds__(256)
qkv_gemm_kernel(const float* __restrict__ X, const float* __restrict__ W,
                const float* __restrict__ bias, unsigned short* __restrict__ Y) {
    const int wave = blockIdx.x * (blockDim.x >> 5) + (threadIdx.x >> 5);
    const int tm = wave / 12, tnb = wave % 12;
    const int lane = threadIdx.x & 31, h = lane >> 4, mn = lane & 15;

    const float4* Af4 = (const float4*)(X + (size_t)(tm * 16 + mn) * DN);

    v8f acc[4];
#pragma unroll
    for (int g = 0; g < 4; ++g) acc[g] = (v8f){};

    for (int k0 = 0; k0 < DN; k0 += 32) {
        const int i1 = (k0 >> 2) + 2 * h;      // run1: k0+8h
        const int i2 = (k0 >> 2) + 4 + 2 * h;  // run2: k0+16+8h
        const v16bf af = frag_from_f32x16(Af4[i1], Af4[i1 + 1], Af4[i2], Af4[i2 + 1]);
#pragma unroll
        for (int g = 0; g < 4; ++g) {
            const float4* Wf4 = (const float4*)(W + (size_t)(tnb * 64 + g * 16 + mn) * DN);
            const int ib = (k0 >> 2) + 2 * h;  // runs: k0+8h and k0+16+8h
            const v16bf bfr = frag_from_f32x16(Wf4[ib], Wf4[ib + 1], Wf4[ib + 4], Wf4[ib + 5]);
            acc[g] = __builtin_amdgcn_wmma_f32_16x16x32_bf16(
                false, af, false, bfr, (short)0, acc[g], false, false);
        }
    }

#pragma unroll
    for (int g = 0; g < 4; ++g) {
        const int col = tnb * 64 + g * 16 + mn;
        const float bv = bias[col];
#pragma unroll
        for (int r = 0; r < 8; ++r) {
            const int row = tm * 16 + r + 8 * h;   // global row = b*1024 + s
            const unsigned short v = f32_to_bf16(acc[g][r] + bv);
            if (TRANSPOSE) {
                const int bb = row >> 10, s = row & 1023;
                Y[(size_t)bb * DN * SN + (size_t)col * SN + s] = v;
            } else {
                Y[(size_t)row * DN + col] = v;
            }
        }
    }
}

// ---------------------------------------------------------------------------
// Row stats: one wave32 per row. {rowmax, sum exp(x-max), rowmin, sum exp(min-x)}
// ---------------------------------------------------------------------------
__global__ void row_stats_kernel(const float* __restrict__ scores,
                                 float4* __restrict__ stats) {
    const int row  = blockIdx.x * (blockDim.x >> 5) + (threadIdx.x >> 5); // < 8192
    const int lane = threadIdx.x & 31;
    const float* rp = scores + (size_t)row * SN;

    float v[32];
    float mx = -3.0e38f, mn = 3.0e38f;
#pragma unroll
    for (int i = 0; i < 32; ++i) {
        v[i] = rp[i * 32 + lane];
        mx = fmaxf(mx, v[i]);
        mn = fminf(mn, v[i]);
    }
#pragma unroll
    for (int o = 16; o > 0; o >>= 1) {
        mx = fmaxf(mx, __shfl_xor(mx, o, 32));
        mn = fminf(mn, __shfl_xor(mn, o, 32));
    }
    float se = 0.f, sn = 0.f;
#pragma unroll
    for (int i = 0; i < 32; ++i) {
        se += __expf(v[i] - mx);
        sn += __expf(mn - v[i]);
    }
#pragma unroll
    for (int o = 16; o > 0; o >>= 1) {
        se += __shfl_xor(se, o, 32);
        sn += __shfl_xor(sn, o, 32);
    }
    if (lane == 0) stats[row] = make_float4(mx, se, mn, sn);
}

// ---------------------------------------------------------------------------
// Radix-select top-k
// ---------------------------------------------------------------------------
__global__ void zero_u32_kernel(unsigned* __restrict__ p, int n) {
    const int i = blockIdx.x * blockDim.x + threadIdx.x;
    if (i < n) p[i] = 0u;
}

__global__ void hist_kernel(const float* __restrict__ scores,
                            const float4* __restrict__ stats,
                            unsigned* __restrict__ hist,
                            const unsigned* __restrict__ state, int pass) {
    const size_t base = ((size_t)blockIdx.x * blockDim.x + threadIdx.x) * 4;
#pragma unroll
    for (int j = 0; j < 4; ++j) {
        const size_t idx = base + j;          // < 8*1024*1024
        const int row = (int)(idx >> 10);     // b*S + q
        const int b   = row >> 10;
        const float4 st = stats[row];
        const float w = __expf(scores[idx] - st.x) / st.y;
        const unsigned bits = __float_as_uint(w);
        if (pass == 0) {
            atomicAdd(&hist[(size_t)b * 65536 + (bits >> 16)], 1u);
        } else if ((bits >> 16) == state[b * 4 + 0]) {
            atomicAdd(&hist[(size_t)b * 65536 + (bits & 0xFFFFu)], 1u);
        }
    }
}

__global__ void select_kernel(const unsigned* __restrict__ hist,
                              unsigned* __restrict__ state, int pass) {
    const int b = blockIdx.x;
    const unsigned* h = hist + (size_t)b * 65536;
    const unsigned target = (pass == 0) ? K_TOPK : state[b * 4 + 1];
    __shared__ unsigned partial[256];
    const int t = threadIdx.x;
    unsigned s = 0;
    for (int i = 0; i < 256; ++i) s += h[65535 - (t * 256 + i)];
    partial[t] = s;
    __syncthreads();
    if (t == 0) {
        unsigned cum = 0, rem = 1; int bin = 0;
        for (int c = 0; c < 256; ++c) {
            if (cum + partial[c] >= target) {
                for (int i = 0; i < 256; ++i) {
                    const int d = c * 256 + i;
                    const unsigned cnt = h[65535 - d];
                    if (cum + cnt >= target) { bin = 65535 - d; rem = target - cum; goto done; }
                    cum += cnt;
                }
            }
            cum += partial[c];
        }
done:
        if (pass == 0) { state[b * 4 + 0] = (unsigned)bin; state[b * 4 + 1] = rem; }
        else           { state[b * 4 + 2] = (state[b * 4 + 0] << 16) | (unsigned)bin; }
    }
}

// ---------------------------------------------------------------------------
// Weights: masked_w (f32 output) + WI (bf16, A of outI) + WV (bf16, A of outV)
// ---------------------------------------------------------------------------
__global__ void weights_kernel(const float* __restrict__ scores,
                               const float4* __restrict__ stats,
                               const unsigned* __restrict__ state,
                               float* __restrict__ masked_w,
                               unsigned short* __restrict__ wi,
                               unsigned short* __restrict__ wv) {
    const size_t base = ((size_t)blockIdx.x * blockDim.x + threadIdx.x) * 4;
#pragma unroll
    for (int j = 0; j < 4; ++j) {
        const size_t idx = base + j;
        const int row = (int)(idx >> 10);
        const int b   = row >> 10;
        const float4 st = stats[row];
        const float sc = scores[idx];
        const float w  = __expf(sc - st.x) / st.y;   // softmax(+)
        const float nw = __expf(st.z - sc) / st.w;   // softmax(-)
        const bool mI = __float_as_uint(w) >= state[b * 4 + 2];
        masked_w[idx] = mI ? w : 0.0f;
        wi[idx]       = f32_to_bf16(mI ? w : 0.0f);
        wv[idx]       = f32_to_bf16(mI ? 0.0f : nw);
    }
}

// ---------------------------------------------------------------------------
// Workspace layout (bytes)
// ---------------------------------------------------------------------------
#define QBF_OFF   ((size_t)0)
#define KBF_OFF   (QBF_OFF + (size_t)BN * SN * DN * 2)
#define VT_OFF    (KBF_OFF + (size_t)BN * SN * DN * 2)       // V^T bf16 [b][d][s]
#define SC_OFF    (VT_OFF  + (size_t)BN * SN * DN * 2)       // f32 scores
#define ST_OFF    (SC_OFF  + (size_t)BN * SN * SN * 4)       // float4 stats
#define HHI_OFF   (ST_OFF  + (size_t)BN * SN * 16)
#define HLO_OFF   (HHI_OFF + (size_t)BN * 65536 * 4)
#define STATE_OFF (HLO_OFF + (size_t)BN * 65536 * 4)
#define WV_OFF    (STATE_OFF + (size_t)256)
#define WI_OFF    (WV_OFF  + (size_t)BN * SN * SN * 2)       // total ~109 MB

extern "C" void kernel_launch(void* const* d_in, const int* in_sizes, int n_in,
                              void* d_out, int out_size, void* d_ws, size_t ws_size,
                              hipStream_t stream) {
    (void)in_sizes; (void)n_in; (void)out_size; (void)ws_size;

    const float* query = (const float*)d_in[0];
    const float* key_  = (const float*)d_in[1];
    const float* value = (const float*)d_in[2];
    const float* Wq = (const float*)d_in[3]; const float* bq = (const float*)d_in[4];
    const float* Wk = (const float*)d_in[5]; const float* bk = (const float*)d_in[6];
    const float* Wv = (const float*)d_in[7]; const float* bv = (const float*)d_in[8];

    char* ws = (char*)d_ws;
    unsigned short* Qbf   = (unsigned short*)(ws + QBF_OFF);
    unsigned short* Kbf   = (unsigned short*)(ws + KBF_OFF);
    unsigned short* Vt    = (unsigned short*)(ws + VT_OFF);
    float*          sc    = (float*)         (ws + SC_OFF);
    float4*         stats = (float4*)        (ws + ST_OFF);
    unsigned*       hHi   = (unsigned*)      (ws + HHI_OFF);
    unsigned*       hLo   = (unsigned*)      (ws + HLO_OFF);
    unsigned*       state = (unsigned*)      (ws + STATE_OFF);
    unsigned short* WVw   = (unsigned short*)(ws + WV_OFF);
    unsigned short* WIw   = (unsigned short*)(ws + WI_OFF);

    float* outI  = (float*)d_out;
    float* outV  = outI + (size_t)BN * SN * DN;
    float* maskw = outI + (size_t)2 * BN * SN * DN;

    const dim3 blk(256);

    // 0) zero histograms + state (contiguous)
    {
        const int n = BN * 65536 * 2 + 64;
        zero_u32_kernel<<<dim3((n + 255) / 256), blk, 0, stream>>>(hHi, n);
    }
    // 1) projections: 512 x 12 blocks of 16x64 per wave -> 768 blocks each
    qkv_gemm_kernel<false><<<dim3(768), blk, 0, stream>>>(query, Wq, bq, Qbf);
    qkv_gemm_kernel<false><<<dim3(768), blk, 0, stream>>>(key_,  Wk, bk, Kbf);
    qkv_gemm_kernel<true ><<<dim3(768), blk, 0, stream>>>(value, Wv, bv, Vt);
    // 2) scores = Q K^T / sqrt(D): 8 batches * 8x8 macro tiles -> 512 blocks
    bf16_gemm_kernel<<<dim3(512), blk, 0, stream>>>(
        Qbf, Kbf, sc, DN, DN, SN, DN, 8, 8,
        (long long)SN * DN, (long long)SN * DN, (long long)SN * SN,
        0.036084391824351614f);
    // 3) row stats
    row_stats_kernel<<<dim3(1024), blk, 0, stream>>>(sc, stats);
    // 4) radix select
    hist_kernel<<<dim3(8192), blk, 0, stream>>>(sc, stats, hHi, state, 0);
    select_kernel<<<dim3(BN), blk, 0, stream>>>(hHi, state, 0);
    hist_kernel<<<dim3(8192), blk, 0, stream>>>(sc, stats, hLo, state, 1);
    select_kernel<<<dim3(BN), blk, 0, stream>>>(hLo, state, 1);
    // 5) weights
    weights_kernel<<<dim3(8192), blk, 0, stream>>>(sc, stats, state, maskw, WIw, WVw);
    // 6) outI = WI @ V ; outV = WV @ V : 8 batches * 8x6 macro tiles -> 384 blocks
    bf16_gemm_kernel<<<dim3(384), blk, 0, stream>>>(
        WIw, Vt, outI, SN, SN, DN, SN, 8, 6,
        (long long)SN * SN, (long long)DN * SN, (long long)SN * DN, 1.0f);
    bf16_gemm_kernel<<<dim3(384), blk, 0, stream>>>(
        WVw, Vt, outV, SN, SN, DN, SN, 8, 6,
        (long long)SN * SN, (long long)DN * SN, (long long)SN * DN, 1.0f);
}
